// SoftLengthRegulator_86294482911719
// MI455X (gfx1250) — matrix operand order repaired and checked by reading, not compiled, not run on hardware
//
#include <hip/hip_runtime.h>
#include <hip/hip_bf16.h>

// ---------------------------------------------------------------------------
// SoftLengthRegulator fused kernels for MI455X (gfx1250, wave32, WMMA)
//   aligned[b,l,d] = sum_t softmax_t(-SIGMA*(l-start[b,t])^2) * x[b,t,d]
//   attn[b,t,l]    = the softmax weights
// Shapes: B=32, T=512, D=512, L=2048 (fixed by the reference harness).
//
// Pipeline:
//   1) slr_scan : per-batch exclusive prefix-sum of durations -> starts (ws)
//   2) slr_xt   : x (B,T,D) f32 -> xT (B,D,T) bf16, one-time transpose (ws)
//   3) slr_main : fused stable softmax (f32) + bf16 WMMA GEMM, 1/Z folded
//                 into the f32 accumulator epilogue. MTILE=64 l-rows/block
//                 halves L2 traffic from xT re-reads; B fragments stream
//                 straight from L2-resident xT (no LDS, no K-loop barriers).
// ---------------------------------------------------------------------------

typedef __attribute__((ext_vector_type(16))) __bf16 v16bf;
typedef __attribute__((ext_vector_type(8)))  __bf16 v8bf;
typedef __attribute__((ext_vector_type(8)))  float  v8f;

#define B_    32
#define T_    512
#define D_    512
#define L_    2048
#define SIGMA 0.2f

#define MTILE 64    // l-rows per block (4 WMMA M-tiles): minimizes xT re-reads
#define PROW  520   // LDS row stride (bf16) of exp tile: 1040B rows (16B aligned, bank stride 4)

// ---------------------------------------------------------------------------
// Kernel 1: per-batch exclusive prefix sum of durations -> starts
// ---------------------------------------------------------------------------
__global__ __launch_bounds__(T_) void slr_scan(const float* __restrict__ dur,
                                               float* __restrict__ starts)
{
    __shared__ float buf[T_];
    const int b = blockIdx.x;
    const int t = threadIdx.x;
    const float d = dur[(size_t)b * T_ + t];
    buf[t] = d;
    __syncthreads();
    for (int off = 1; off < T_; off <<= 1) {
        float a = buf[t];
        float c = (t >= off) ? buf[t - off] : 0.0f;
        __syncthreads();
        buf[t] = a + c;
        __syncthreads();
    }
    starts[(size_t)b * T_ + t] = buf[t] - d;
}

// ---------------------------------------------------------------------------
// Kernel 2: one-time transpose + downconvert: xT[b][d][t] = bf16(x[b][t][d])
// ---------------------------------------------------------------------------
__global__ __launch_bounds__(256) void slr_xt(const float* __restrict__ x,
                                              __bf16* __restrict__ xT)
{
    __shared__ __bf16 tile[32][34];
    const int b  = blockIdx.z;
    const int t0 = blockIdx.x * 32;
    const int d0 = blockIdx.y * 32;
    const int tx = threadIdx.x & 31;
    const int ty = threadIdx.x >> 5;
    #pragma unroll
    for (int i = 0; i < 4; ++i) {
        const int t = ty + i * 8;
        tile[t][tx] = (__bf16)x[((size_t)b * T_ + t0 + t) * D_ + d0 + tx];
    }
    __syncthreads();
    #pragma unroll
    for (int i = 0; i < 4; ++i) {
        const int d = ty + i * 8;
        xT[((size_t)b * D_ + d0 + d) * T_ + t0 + tx] = tile[tx][d];
    }
}

// ---------------------------------------------------------------------------
// Kernel 3: fused softmax + WMMA GEMM.
// Grid: (L/MTILE, B). Block: 256 threads = 8 waves.
// Block: 64 l-rows of batch b. Wave w owns d in [w*64, w*64+64) (4 N-tiles)
// and all 4 M-tiles -> 16 f32 accumulator tiles per wave; each B fragment
// feeds 4 WMMAs.
// ---------------------------------------------------------------------------
__global__ __launch_bounds__(256) void slr_main(const __bf16* __restrict__ xT,
                                                const float*  __restrict__ starts,
                                                float* __restrict__ aligned,
                                                float* __restrict__ attn)
{
    __shared__ __align__(16) float  s_start[T_];          //  2048 B
    __shared__ __align__(16) __bf16 s_p[MTILE * PROW];    // 66560 B  exp tile [l_row][t]
    __shared__ __align__(16) float  s_red[256];           //  1024 B
    __shared__ __align__(16) float  s_rcpZ[MTILE];        //   256 B

    const int tid = threadIdx.x;
    const int b   = blockIdx.y;
    const int l0  = blockIdx.x * MTILE;

    for (int i = tid; i < T_; i += 256) s_start[i] = starts[(size_t)b * T_ + i];
    __syncthreads();

    // ---- stable softmax, two groups of 32 rows (8 threads/row, 64 t each) ----
    const int rbase = tid >> 3;              // 0..31
    const int tx    = tid & 7;
    const int tbase = tx * 64;

    for (int g = 0; g < 2; ++g) {
        const int   r  = g * 32 + rbase;     // l-row within block
        const float lf = (float)(l0 + r);

        // pass 1: min squared distance (negated logit max)
        float mind2 = 3.4e38f;
        #pragma unroll
        for (int j = 0; j < 64; ++j) {
            float s = lf - s_start[tbase + j];
            mind2 = fminf(mind2, s * s);
        }
        s_red[tid] = mind2;
        __syncthreads();
        if (tx == 0) {
            float m = s_red[tid];
            #pragma unroll
            for (int i = 1; i < 8; ++i) m = fminf(m, s_red[tid + i]);
            s_red[tid] = m;
        }
        __syncthreads();
        mind2 = s_red[rbase * 8];
        __syncthreads();

        // pass 2: exp + row sum (f32 in registers)
        float e[64];
        float lsum = 0.0f;
        #pragma unroll
        for (int j = 0; j < 64; ++j) {
            float s = lf - s_start[tbase + j];
            float v = __expf(-SIGMA * (s * s - mind2));
            e[j] = v;
            lsum += v;
        }
        s_red[tid] = lsum;
        __syncthreads();
        if (tx == 0) {
            float z = s_red[tid];
            #pragma unroll
            for (int i = 1; i < 8; ++i) z += s_red[tid + i];
            s_rcpZ[r] = 1.0f / z;
        }
        __syncthreads();

        // write attn (f32, exact); stash raw exp as packed bf16 for WMMA A
        const float rz = s_rcpZ[r];
        float* arow = attn + (size_t)b * T_ * L_ + (size_t)(l0 + r);
        #pragma unroll
        for (int jj = 0; jj < 8; ++jj) {
            v8bf pk;
            #pragma unroll
            for (int i = 0; i < 8; ++i) {
                const int j = jj * 8 + i;
                arow[(size_t)(tbase + j) * L_] = e[j] * rz;  // normalized, f32
                pk[i] = (__bf16)e[j];                        // raw exp (<=1)
            }
            *(v8bf*)&s_p[r * PROW + tbase + jj * 8] = pk;    // 16B LDS store
        }
        __syncthreads();
    }

    // ---- WMMA GEMM: C[64 x 512] = P[64 x 512] * X[512 x 512] ----
    const int wave   = tid >> 5;        // 0..7
    const int lane   = tid & 31;
    const int hf     = lane >> 4;       // half-wave
    const int ln     = lane & 15;       // M (A) / N (B,C,D) index
    const int dBase0 = wave * 64;

    const __bf16* xTb = xT + (size_t)b * D_ * T_;

    v8f acc[16];                         // [mt*4 + nt]
    #pragma unroll
    for (int c = 0; c < 16; ++c)
        #pragma unroll
        for (int i = 0; i < 8; ++i) acc[c][i] = 0.0f;

    for (int ks = 0; ks < 16; ++ks) {
        const int t0 = ks * 32;

        // A fragments (16x32 bf16 each): lanes 0-15 -> K {0..7,16..23}; +8 for hf=1
        v16bf af[4];
        #pragma unroll
        for (int mt = 0; mt < 4; ++mt) {
            const int row = mt * 16 + ln;
            const v8bf a0 = *(const v8bf*)&s_p[row * PROW + t0 + hf * 8];
            const v8bf a1 = *(const v8bf*)&s_p[row * PROW + t0 + 16 + hf * 8];
            #pragma unroll
            for (int i = 0; i < 8; ++i) { af[mt][i] = a0[i]; af[mt][i + 8] = a1[i]; }
        }

        // B fragments from L2-resident xT; each feeds 4 WMMAs.
        #pragma unroll
        for (int nt = 0; nt < 4; ++nt) {
            const __bf16* bp = xTb + (size_t)(dBase0 + nt * 16 + ln) * T_ + t0 + hf * 8;
            v16bf bf;
            const v8bf b0 = *(const v8bf*)bp;
            const v8bf b1 = *(const v8bf*)(bp + 16);
            #pragma unroll
            for (int i = 0; i < 8; ++i) { bf[i] = b0[i]; bf[i + 8] = b1[i]; }

            #pragma unroll
            for (int mt = 0; mt < 4; ++mt) {
                acc[mt * 4 + nt] = __builtin_amdgcn_wmma_f32_16x16x32_bf16(
                    false, af[mt], false, bf, (short)0, acc[mt * 4 + nt], false, false);
            }
        }
    }

    // ---- epilogue: fold 1/Z row-wise, store aligned (f32) ----
    // C/D layout: VGPR vr -> row m = vr + 8*hf ; column n = ln
    #pragma unroll
    for (int mt = 0; mt < 4; ++mt) {
        #pragma unroll
        for (int vr = 0; vr < 8; ++vr) {
            const int   m  = mt * 16 + vr + 8 * hf;
            const float sc = s_rcpZ[m];
            float* orow = aligned + ((size_t)b * L_ + (l0 + m)) * D_;
            #pragma unroll
            for (int nt = 0; nt < 4; ++nt)
                orow[dBase0 + nt * 16 + ln] = acc[mt * 4 + nt][vr] * sc;
        }
    }
}

// ---------------------------------------------------------------------------
extern "C" void kernel_launch(void* const* d_in, const int* in_sizes, int n_in,
                              void* d_out, int out_size, void* d_ws, size_t ws_size,
                              hipStream_t stream)
{
    (void)in_sizes; (void)n_in; (void)out_size; (void)ws_size;

    const float* x   = (const float*)d_in[0];   // (B, T, D) f32
    const float* dur = (const float*)d_in[1];   // (B, T)    f32
    // d_in[2] = max_length scalar (2048), hardcoded above.

    float* aligned = (float*)d_out;                              // (B, L, D)
    float* attn    = (float*)d_out + (size_t)B_ * L_ * D_;       // (B, T, L)

    float*  starts = (float*)d_ws;                               // 64 KB
    __bf16* xT     = (__bf16*)((char*)d_ws + (size_t)B_ * T_ * sizeof(float)); // 16.8 MB

    slr_scan<<<dim3(B_), dim3(T_), 0, stream>>>(dur, starts);
    slr_xt  <<<dim3(T_ / 32, D_ / 32, B_), dim3(256), 0, stream>>>(x, xT);
    slr_main<<<dim3(L_ / MTILE, B_), dim3(256), 0, stream>>>(xT, starts, aligned, attn);
}
